// SFA_2894807957859
// MI455X (gfx1250) — compile-verified, matrix operand order checked by reading, not run on hardware
//
#include <hip/hip_runtime.h>
#include <cstdint>

// ---------------------------------------------------------------------------
// MI455X (gfx1250) implementation of the SFA block.
// All matmuls use v_wmma_f32_16x16x32_bf16 (wave32, 8 waves / 256-thread WG).
// GEMM tile staging uses GLOBAL_LOAD_ASYNC_TO_LDS_B128 (ASYNCcnt) when the
// toolchain exposes the builtin; falls back to load+ds_store otherwise.
// Assumes ws_size >= ~1.1 GB (all intermediates live in d_ws).
// ---------------------------------------------------------------------------

typedef unsigned short bfu;                                       // raw bf16
typedef __bf16  bf16_t;
typedef __attribute__((ext_vector_type(16))) bf16_t v16bf;
typedef __attribute__((ext_vector_type(8)))  float  v8f;

#define EPS_ 1e-5f

#if defined(__AMDGCN__) && __has_builtin(__builtin_amdgcn_global_load_async_to_lds_b128) && __has_builtin(__builtin_amdgcn_s_wait_asynccnt)
#define USE_ASYNC_LDS 1
#define AS1 __attribute__((address_space(1)))
#define AS3 __attribute__((address_space(3)))
typedef int v4i __attribute__((vector_size(4 * sizeof(int))));
__device__ __forceinline__ void async_cp16(const bfu* g, bfu* l) {
  __builtin_amdgcn_global_load_async_to_lds_b128(
      (AS1 v4i*)(void*)const_cast<bfu*>(g), (AS3 v4i*)(void*)l, 0, 0);
}
__device__ __forceinline__ void async_wait0() {
  __builtin_amdgcn_s_wait_asynccnt(0);
}
#endif

__device__ __forceinline__ float bf2f(bfu u) {
  union { unsigned int i; float f; } c; c.i = ((unsigned int)u) << 16; return c.f;
}
__device__ __forceinline__ bfu f2bf(float f) {
  union { float f; unsigned int i; } c; c.f = f;
  unsigned int x = c.i;
  unsigned int r = x + 0x7fffu + ((x >> 16) & 1u);   // RNE
  return (bfu)(r >> 16);
}
__device__ __forceinline__ v8f vzero8() {
  v8f z;
#pragma unroll
  for (int i = 0; i < 8; ++i) z[i] = 0.0f;
  return z;
}
__device__ __forceinline__ void fill_dct(float* D, int t) {
  if (t < 64) {
    int u = t >> 3, x = t & 7;
    float s = (u == 0) ? 0.35355339059327373f : 0.5f;   // sqrt(1/8), sqrt(2/8)
    D[t] = s * __cosf((2.0f * x + 1.0f) * (float)u * 3.14159265358979323846f / 16.0f);
  }
}

// ------------------------- small elementwise kernels ------------------------

__global__ void k_f2bf(const float* __restrict__ s, bfu* __restrict__ d, long long n) {
  long long i = (long long)blockIdx.x * 256 + threadIdx.x;
  long long st = (long long)gridDim.x * 256;
  for (; i < n; i += st) d[i] = f2bf(s[i]);
}

__global__ void k_zero16(bfu* __restrict__ p, long long n) {
  long long i = (long long)blockIdx.x * 256 + threadIdx.x;
  long long st = (long long)gridDim.x * 256;
  for (; i < n; i += st) p[i] = 0;
}

// split conv2 weight [o][i][ky][kx] -> 9 tap matrices wt[tap][o*1024+i] (bf16)
__global__ void k_taps(const float* __restrict__ c2, bfu* __restrict__ wt) {
  long long n = 1024ll * 1024ll * 9ll;
  long long i = (long long)blockIdx.x * 256 + threadIdx.x;
  long long st = (long long)gridDim.x * 256;
  for (; i < n; i += st) {
    int tap = (int)(i % 9);
    long long oi = i / 9;
    wt[(long long)tap * 1048576ll + oi] = f2bf(c2[oi * 9 + tap]);
  }
}

// ----------------------------- block DCT (8x8) ------------------------------

__global__ __launch_bounds__(256) void k_dct(const float* __restrict__ feat,
                                             float* __restrict__ out) {
  __shared__ float D[64];
  int t = threadIdx.x;
  fill_dct(D, t);
  __syncthreads();
  long long gid = (long long)blockIdx.x * 256 + t;        // 8*1024*64 blocks
  int plane = (int)(gid >> 6);
  int blk   = (int)(gid & 63);
  size_t base = (size_t)plane * 4096 + (size_t)(blk >> 3) * 512 + (size_t)(blk & 7) * 8;
  float P[8][8];
#pragma unroll
  for (int x = 0; x < 8; ++x)
#pragma unroll
    for (int y = 0; y < 8; ++y) P[x][y] = feat[base + x * 64 + y];
#pragma unroll
  for (int u = 0; u < 8; ++u) {
    float tm[8];
#pragma unroll
    for (int y = 0; y < 8; ++y) {
      float s = 0.f;
#pragma unroll
      for (int x = 0; x < 8; ++x) s += D[u * 8 + x] * P[x][y];
      tm[y] = s;
    }
#pragma unroll
    for (int v = 0; v < 8; ++v) {
      float s = 0.f;
#pragma unroll
      for (int y = 0; y < 8; ++y) s += tm[y] * D[v * 8 + y];
      out[base + u * 64 + v] = s;
    }
  }
}

// ------------- channel LayerNorm (both branches) + window layout ------------
// one wave per pixel; outputs fw/sw as (win, n, c) bf16

__global__ __launch_bounds__(256) void k_lnwin(const float* __restrict__ feat,
                                               const float* __restrict__ dct,
                                               const float* __restrict__ n1w, const float* __restrict__ n1b,
                                               const float* __restrict__ n2w, const float* __restrict__ n2b,
                                               bfu* __restrict__ fw, bfu* __restrict__ sw) {
  int t = threadIdx.x, wv = t >> 5, lid = t & 31;
  int p = blockIdx.x * 8 + wv;                          // 32768 pixels
  int b = p >> 12, rem = p & 4095;
  int Y = rem >> 6, X = rem & 63;
  size_t base = (size_t)b * (1024ull * 4096ull) + (size_t)rem;
  float av[32], dv[32];
  float s1 = 0.f, q1 = 0.f, s2 = 0.f, q2 = 0.f;
#pragma unroll
  for (int j = 0; j < 32; ++j) {
    int c = lid + j * 32;
    float fv = feat[base + (size_t)c * 4096];
    float dd = dct [base + (size_t)c * 4096];
    av[j] = fv; dv[j] = dd;
    s1 += fv; q1 += fv * fv; s2 += dd; q2 += dd * dd;
  }
#pragma unroll
  for (int m = 16; m > 0; m >>= 1) {
    s1 += __shfl_xor(s1, m, 32); q1 += __shfl_xor(q1, m, 32);
    s2 += __shfl_xor(s2, m, 32); q2 += __shfl_xor(q2, m, 32);
  }
  float mu1 = s1 * (1.f / 1024.f), va1 = q1 * (1.f / 1024.f) - mu1 * mu1;
  float mu2 = s2 * (1.f / 1024.f), va2 = q2 * (1.f / 1024.f) - mu2 * mu2;
  float r1 = rsqrtf(va1 + EPS_), r2 = rsqrtf(va2 + EPS_);
  int win = b * 64 + (Y >> 3) * 8 + (X >> 3);
  int n   = (Y & 7) * 8 + (X & 7);
  size_t ob = ((size_t)win * 64 + n) * 1024;
#pragma unroll
  for (int j = 0; j < 32; ++j) {
    int c = lid + j * 32;
    sw[ob + c] = f2bf((av[j] - mu1) * r1 * n2w[c] + n2b[c]);
    fw[ob + c] = f2bf((dv[j] - mu2) * r2 * n1w[c] + n1b[c]);
  }
}

// ------------------------------ WMMA GEMM ----------------------------------
// O[M,N] = A[M,K] * W[N,K]^T (+bias)(+addsrc)(+gelu) ; bf16 in, f32 acc.
// amode: 0 dense (lda=K) ; 1 read from zero-padded NHWC y1p with tap (dy,dx)
// omode: 0 f32 ; 1 bf16 ; 2 gelu->bf16 ; 3 f32 += addsrc ; 4 bf16 into padded y1p

#define AST 56   // 32 + 24 pad (112 B rows: 16B aligned, conflict-free steps)

__global__ __launch_bounds__(256)
void k_gemm(const bfu* __restrict__ A, const bfu* __restrict__ W,
            const float* __restrict__ bias, const float* __restrict__ addsrc,
            float* __restrict__ outf, bfu* __restrict__ outb,
            int M, int N, int K, int amode, int dy, int dx, int omode) {
  __shared__ bfu As[128 * AST];
  __shared__ bfu Bs[128 * AST];
  int t = threadIdx.x;
  int wv = t >> 5, lid = t & 31, hf = lid >> 4, r = lid & 15;
  int wm = wv >> 2, wn = wv & 3;                 // 2 x 4 wave grid
  int m0 = blockIdx.y * 128, n0 = blockIdx.x * 128;
  v8f acc[4][2];
#pragma unroll
  for (int a = 0; a < 4; ++a)
#pragma unroll
    for (int b = 0; b < 2; ++b) acc[a][b] = vzero8();

  for (int k0 = 0; k0 < K; k0 += 32) {
#pragma unroll
    for (int j = 0; j < 2; ++j) {
      int job = t + j * 256;
      int row = job >> 2, col = (job & 3) * 8;
      int gm = m0 + row;
      size_t aoff;
      if (amode == 0) {
        aoff = (size_t)gm * (size_t)K + (size_t)(k0 + col);
      } else {
        int bb = gm >> 12, rem = gm & 4095, y = rem >> 6, x = rem & 63;
        aoff = ((size_t)(bb * 66 + y + dy) * 66 + (size_t)(x + dx)) * 1024 + (size_t)(k0 + col);
      }
      int gn = n0 + row;
      size_t boff = (size_t)gn * (size_t)K + (size_t)(k0 + col);
#if defined(USE_ASYNC_LDS)
      async_cp16(&A[aoff], &As[row * AST + col]);     // global_load_async_to_lds_b128
      async_cp16(&W[boff], &Bs[row * AST + col]);
#else
      *(uint4*)(&As[row * AST + col]) = *(const uint4*)(&A[aoff]);
      *(uint4*)(&Bs[row * AST + col]) = *(const uint4*)(&W[boff]);
#endif
      if (k0 + 32 < K) __builtin_prefetch(&A[aoff + 32], 0, 1);   // global_prefetch_b8
    }
#if defined(USE_ASYNC_LDS)
    async_wait0();                                    // s_wait_asynccnt 0
#endif
    __syncthreads();
    union Fr { v16bf v; uint4 q[2]; };
    Fr fa[4], fb[2];
#pragma unroll
    for (int mi = 0; mi < 4; ++mi) {
      const bfu* p = &As[(wm * 64 + mi * 16 + r) * AST];
      fa[mi].q[0] = *(const uint4*)(p + 8 * hf);
      fa[mi].q[1] = *(const uint4*)(p + 16 + 8 * hf);
    }
#pragma unroll
    for (int ni = 0; ni < 2; ++ni) {
      const bfu* p = &Bs[(wn * 32 + ni * 16 + r) * AST];
      fb[ni].q[0] = *(const uint4*)(p + 8 * hf);
      fb[ni].q[1] = *(const uint4*)(p + 16 + 8 * hf);
    }
#pragma unroll
    for (int mi = 0; mi < 4; ++mi)
#pragma unroll
      for (int ni = 0; ni < 2; ++ni)
        acc[mi][ni] = __builtin_amdgcn_wmma_f32_16x16x32_bf16(
            false, fa[mi].v, false, fb[ni].v, (short)0, acc[mi][ni], false, false);
    __syncthreads();
  }
  // epilogue: D layout -> row = i + 8*hf, col = lane r
#pragma unroll
  for (int mi = 0; mi < 4; ++mi) {
#pragma unroll
    for (int ni = 0; ni < 2; ++ni) {
      int col = n0 + wn * 32 + ni * 16 + r;
#pragma unroll
      for (int i = 0; i < 8; ++i) {
        int gm = m0 + wm * 64 + mi * 16 + i + 8 * hf;
        float v = acc[mi][ni][i];
        if (bias) v += bias[col];
        if (omode == 2) v = 0.5f * v * (1.0f + erff(v * 0.70710678118654752f));
        if (omode == 3) v += addsrc[(size_t)gm * (size_t)N + col];
        if (omode == 0 || omode == 3) {
          outf[(size_t)gm * (size_t)N + col] = v;
        } else if (omode == 4) {
          int bb = gm >> 12, rem = gm & 4095, y = rem >> 6, x = rem & 63;
          outb[((size_t)(bb * 66 + y + 1) * 66 + (size_t)(x + 1)) * 1024 + col] = f2bf(v);
        } else {
          outb[(size_t)gm * (size_t)N + col] = f2bf(v);
        }
      }
    }
  }
}

// ------------------------- fused window attention ---------------------------
// one WG per (window, head): S = (Q*scale) K^T + relpos-bias, softmax, P V.

__global__ __launch_bounds__(256)
void k_attn(const bfu* __restrict__ q, const bfu* __restrict__ k, const bfu* __restrict__ v,
            const float* __restrict__ table, bfu* __restrict__ ctx) {
  __shared__ __align__(16) unsigned char smraw[61440];
  bfu*   qh = (bfu*)smraw;                  // 64 x 152 bf16 (19456 B)
  float* sm = (float*)smraw;                // aliases qh: 64 x 66 f32 (16896 B)
  bfu*   kh = (bfu*)(smraw + 19456);        // 64 x 152 bf16
  bfu*   pm = (bfu*)(smraw + 19456);        // aliases kh: 64 x 88 bf16 (11264 B)
  bfu*   vt = (bfu*)(smraw + 38912);        // 128 x 88 bf16 (V transposed)

  int t = threadIdx.x;
  int win = blockIdx.x >> 3, head = blockIdx.x & 7;
  size_t gb = ((size_t)win * 64) * 1024 + (size_t)head * 128;

#if defined(USE_ASYNC_LDS)
  // K rows are contiguous 256 B: stage via async-to-LDS (16 B chunks)
  for (int idx = t; idx < 1024; idx += 256) {
    int i = idx >> 4, d = (idx & 15) * 8;
    async_cp16(&k[gb + (size_t)i * 1024 + d], &kh[i * 152 + d]);
  }
#endif
  for (int idx = t; idx < 8192; idx += 256) {
    int i = idx >> 7, d = idx & 127;
    size_t g = gb + (size_t)i * 1024 + d;
    qh[i * 152 + d] = f2bf(bf2f(q[g]) * 0.08838834764831845f);   // HD^-0.5
#if !defined(USE_ASYNC_LDS)
    kh[i * 152 + d] = k[g];
#endif
    vt[d * 88 + i]  = v[g];
  }
#if defined(USE_ASYNC_LDS)
  async_wait0();
#endif
  __syncthreads();

  int wv = t >> 5, lid = t & 31, hf = lid >> 4, r = lid & 15;
  union Fr { v16bf v; uint4 q[2]; };

  // ---- S = Q K^T : waves 4(M) x 2(N-halves), 2 acc tiles each
  int mi = wv >> 1, nj = wv & 1;
  v8f a0 = vzero8(), a1 = vzero8();
#pragma unroll
  for (int kk = 0; kk < 4; ++kk) {
    Fr fa, f0, f1;
    const bfu* pa = &qh[(mi * 16 + r) * 152 + kk * 32];
    fa.q[0] = *(const uint4*)(pa + 8 * hf);
    fa.q[1] = *(const uint4*)(pa + 16 + 8 * hf);
    const bfu* p0 = &kh[(nj * 32 + r) * 152 + kk * 32];
    f0.q[0] = *(const uint4*)(p0 + 8 * hf);
    f0.q[1] = *(const uint4*)(p0 + 16 + 8 * hf);
    const bfu* p1 = &kh[(nj * 32 + 16 + r) * 152 + kk * 32];
    f1.q[0] = *(const uint4*)(p1 + 8 * hf);
    f1.q[1] = *(const uint4*)(p1 + 16 + 8 * hf);
    a0 = __builtin_amdgcn_wmma_f32_16x16x32_bf16(false, fa.v, false, f0.v, (short)0, a0, false, false);
    a1 = __builtin_amdgcn_wmma_f32_16x16x32_bf16(false, fa.v, false, f1.v, (short)0, a1, false, false);
  }
  __syncthreads();                       // qh reads done -> sm may alias
#pragma unroll
  for (int i = 0; i < 8; ++i) {
    int row = mi * 16 + i + 8 * hf;
    sm[row * 66 + nj * 32 + r]      = a0[i];
    sm[row * 66 + nj * 32 + 16 + r] = a1[i];
  }
  __syncthreads();

  // ---- bias + softmax (one thread per row)
  if (t < 64) {
    int row = t;
    int ri = 15 * (row >> 3) + (row & 7);
    float mx = -1e30f;
    for (int j = 0; j < 64; ++j) {
      int jj = 63 - j;
      int rj = 15 * (jj >> 3) + (jj & 7);
      float x = sm[row * 66 + j] + table[(ri + rj) * 8 + head];
      sm[row * 66 + j] = x;
      mx = fmaxf(mx, x);
    }
    float s = 0.f;
    for (int j = 0; j < 64; ++j) s += __expf(sm[row * 66 + j] - mx);
    float inv = 1.0f / s;
    for (int j = 0; j < 64; ++j)
      pm[row * 88 + j] = f2bf(__expf(sm[row * 66 + j] - mx) * inv);
  }
  __syncthreads();

  // ---- O = P V : waves 2(M) x 4(N), 2x2 acc tiles each
  int wm = wv >> 2, wn = wv & 3;
  v8f acc[2][2];
#pragma unroll
  for (int a = 0; a < 2; ++a)
#pragma unroll
    for (int b = 0; b < 2; ++b) acc[a][b] = vzero8();
#pragma unroll
  for (int jj = 0; jj < 2; ++jj) {
    Fr fa[2], fb[2];
#pragma unroll
    for (int m2 = 0; m2 < 2; ++m2) {
      const bfu* p = &pm[(wm * 32 + m2 * 16 + r) * 88 + jj * 32];
      fa[m2].q[0] = *(const uint4*)(p + 8 * hf);
      fa[m2].q[1] = *(const uint4*)(p + 16 + 8 * hf);
    }
#pragma unroll
    for (int n2 = 0; n2 < 2; ++n2) {
      const bfu* p = &vt[(wn * 32 + n2 * 16 + r) * 88 + jj * 32];
      fb[n2].q[0] = *(const uint4*)(p + 8 * hf);
      fb[n2].q[1] = *(const uint4*)(p + 16 + 8 * hf);
    }
#pragma unroll
    for (int m2 = 0; m2 < 2; ++m2)
#pragma unroll
      for (int n2 = 0; n2 < 2; ++n2)
        acc[m2][n2] = __builtin_amdgcn_wmma_f32_16x16x32_bf16(
            false, fa[m2].v, false, fb[n2].v, (short)0, acc[m2][n2], false, false);
  }
#pragma unroll
  for (int m2 = 0; m2 < 2; ++m2)
#pragma unroll
    for (int n2 = 0; n2 < 2; ++n2)
#pragma unroll
      for (int i = 0; i < 8; ++i) {
        int row = wm * 32 + m2 * 16 + i + 8 * hf;
        int col = wn * 32 + n2 * 16 + r;
        ctx[((size_t)win * 64 + row) * 1024 + (size_t)head * 128 + col] = f2bf(acc[m2][n2][i]);
      }
}

// ----------- windows -> NHWC, with inverse block-DCT (stf branch) ----------

__global__ __launch_bounds__(256) void k_idct_cat(const float* __restrict__ owin,
                                                  bfu* __restrict__ cat) {
  __shared__ float D[64];
  int t = threadIdx.x;
  fill_dct(D, t);
  __syncthreads();
  long long gid = (long long)blockIdx.x * 256 + t;   // 512 * 1024
  int win = (int)(gid >> 10), c = (int)(gid & 1023);
  int b = win >> 6, by = (win >> 3) & 7, bx = win & 7;
  float Cf[8][8];
  size_t ib = ((size_t)win * 64) * 1024 + c;
#pragma unroll
  for (int u = 0; u < 8; ++u)
#pragma unroll
    for (int v = 0; v < 8; ++v) Cf[u][v] = owin[ib + (size_t)(u * 8 + v) * 1024];
#pragma unroll
  for (int x = 0; x < 8; ++x) {
    float tv[8];
#pragma unroll
    for (int v = 0; v < 8; ++v) {
      float s = 0.f;
#pragma unroll
      for (int u = 0; u < 8; ++u) s += D[u * 8 + x] * Cf[u][v];
      tv[v] = s;
    }
#pragma unroll
    for (int y = 0; y < 8; ++y) {
      float s = 0.f;
#pragma unroll
      for (int v = 0; v < 8; ++v) s += tv[v] * D[v * 8 + y];
      int Y = by * 8 + x, X = bx * 8 + y;
      cat[(((size_t)b * 4096) + (size_t)Y * 64 + X) * 2048 + c] = f2bf(s);
    }
  }
}

// windows -> NHWC copy into cols [1024,2048) of cat (fts branch)
__global__ void k_win2cat(const float* __restrict__ owin, bfu* __restrict__ cat) {
  long long i = (long long)blockIdx.x * 256 + threadIdx.x;    // 33554432 exact
  int c = (int)(i & 1023);
  long long mn = i >> 10;
  int n = (int)(mn & 63), win = (int)(mn >> 6);
  int b = win >> 6, by = (win >> 3) & 7, bx = win & 7;
  int u = n >> 3, vv = n & 7;
  int Y = by * 8 + u, X = bx * 8 + vv;
  cat[(((size_t)b * 4096) + (size_t)Y * 64 + X) * 2048 + 1024 + c] = f2bf(owin[i]);
}

// --------------------------- BatchNorm (training) ---------------------------

__global__ __launch_bounds__(256) void k_bnstats(const float* __restrict__ y,
                                                 float* __restrict__ mean, float* __restrict__ istd) {
  int c = blockIdx.x, t = threadIdx.x;
  float s = 0.f, qq = 0.f;
  for (int j = t; j < 32768; j += 256) {
    float v = y[(size_t)j * 1024 + c];
    s += v; qq += v * v;
  }
  __shared__ float ss[256], sq[256];
  ss[t] = s; sq[t] = qq;
  __syncthreads();
  for (int m = 128; m > 0; m >>= 1) {
    if (t < m) { ss[t] += ss[t + m]; sq[t] += sq[t + m]; }
    __syncthreads();
  }
  if (t == 0) {
    float mu = ss[0] * (1.f / 32768.f);
    float va = sq[0] * (1.f / 32768.f) - mu * mu;
    mean[c] = mu; istd[c] = rsqrtf(va + EPS_);
  }
}

__global__ void k_bnres(const float* __restrict__ y, const float* __restrict__ mean,
                        const float* __restrict__ istd, const float* __restrict__ g,
                        const float* __restrict__ b2, const float* __restrict__ feat,
                        float* __restrict__ res) {
  long long i = (long long)blockIdx.x * 256 + threadIdx.x;    // 33554432 exact
  int c = (int)(i & 1023);
  long long m = i >> 10;
  int bb = (int)(m >> 12), pp = (int)(m & 4095);
  float v = (y[i] - mean[c]) * istd[c] * g[c] + b2[c];
  v = fmaxf(v, 0.f);
  res[i] = feat[((size_t)bb * 1024 + c) * 4096 + pp] + v;
}

// LayerNorm over last dim of res (NHWC rows), bf16 out for the MLP
__global__ __launch_bounds__(256) void k_ln3(const float* __restrict__ res,
                                             const float* __restrict__ w, const float* __restrict__ b,
                                             bfu* __restrict__ tn) {
  int t = threadIdx.x, wv = t >> 5, lid = t & 31;
  int m = blockIdx.x * 8 + wv;
  float a[32]; float s = 0.f, qq = 0.f;
#pragma unroll
  for (int j = 0; j < 32; ++j) {
    float v = res[(size_t)m * 1024 + lid + j * 32];
    a[j] = v; s += v; qq += v * v;
  }
#pragma unroll
  for (int x = 16; x > 0; x >>= 1) { s += __shfl_xor(s, x, 32); qq += __shfl_xor(qq, x, 32); }
  float mu = s * (1.f / 1024.f);
  float r = rsqrtf(qq * (1.f / 1024.f) - mu * mu + EPS_);
#pragma unroll
  for (int j = 0; j < 32; ++j) {
    int c = lid + j * 32;
    tn[(size_t)m * 1024 + c] = f2bf((a[j] - mu) * r * w[c] + b[c]);
  }
}

__global__ void k_hwc2chw(const float* __restrict__ fin, float* __restrict__ out) {
  long long i = (long long)blockIdx.x * 256 + threadIdx.x;    // 33554432 exact
  int c = (int)(i & 1023);
  long long m = i >> 10;
  int bb = (int)(m >> 12), pp = (int)(m & 4095);
  out[((size_t)bb * 1024 + c) * 4096 + pp] = fin[i];
}

// ------------------------------- launcher ----------------------------------

extern "C" void kernel_launch(void* const* d_in, const int* in_sizes, int n_in,
                              void* d_out, int out_size, void* d_ws, size_t ws_size,
                              hipStream_t stream) {
  (void)in_sizes; (void)n_in; (void)out_size; (void)ws_size;
  const float* feat  = (const float*)d_in[0];
  const float* wq1 = (const float*)d_in[2],  *wk1 = (const float*)d_in[3];
  const float* wv1 = (const float*)d_in[4],  *wo1 = (const float*)d_in[5];
  const float* bq1 = (const float*)d_in[6],  *bk1 = (const float*)d_in[7];
  const float* bv1 = (const float*)d_in[8],  *bo1 = (const float*)d_in[9];
  const float* tb1 = (const float*)d_in[10];
  const float* wq2 = (const float*)d_in[11], *wk2 = (const float*)d_in[12];
  const float* wv2 = (const float*)d_in[13], *wo2 = (const float*)d_in[14];
  const float* bq2 = (const float*)d_in[15], *bk2 = (const float*)d_in[16];
  const float* bv2 = (const float*)d_in[17], *bo2 = (const float*)d_in[18];
  const float* tb2 = (const float*)d_in[19];
  const float* n1w = (const float*)d_in[20], *n1b = (const float*)d_in[21];
  const float* n2w = (const float*)d_in[22], *n2b = (const float*)d_in[23];
  const float* n3w = (const float*)d_in[24], *n3b = (const float*)d_in[25];
  const float* c1w = (const float*)d_in[26], *c1b = (const float*)d_in[27];
  const float* c2w = (const float*)d_in[28], *c2b = (const float*)d_in[29];
  const float* bng = (const float*)d_in[30], *bnb = (const float*)d_in[31];
  const float* f1w = (const float*)d_in[32], *f1b = (const float*)d_in[33];
  const float* f2w = (const float*)d_in[34], *f2b = (const float*)d_in[35];

  char* ws = (char*)d_ws;
  size_t off = 0;
  auto take = [&](size_t bytes) -> char* {
    char* p = ws + off;
    off += (bytes + 255) & ~(size_t)255;
    return p;
  };
  float* dctb = (float*)take(134217728);          // DCT coeffs; later reused as owin
  bfu*   fwb  = (bfu*)take(67108864);
  bfu*   swb  = (bfu*)take(67108864);
  bfu*   qb   = (bfu*)take(67108864);
  bfu*   kb   = (bfu*)take(67108864);
  bfu*   vb   = (bfu*)take(67108864);
  bfu*   ctx  = (bfu*)take(67108864);
  bfu*   cat  = (bfu*)take(134217728);            // concat NHWC; later reused as MLP hidden
  bfu*   y1p  = (bfu*)take(71368704);             // 8 x 66 x 66 x 1024 bf16 (zero padded)
  float* y2   = (float*)take(134217728);          // conv2 out; later reused as final NHWC
  float* res  = (float*)take(134217728);
  bfu*   tnb  = (bfu*)take(67108864);
  float* bmean = (float*)take(4096);
  float* bistd = (float*)take(4096);
  bfu* wq1b = (bfu*)take(2097152); bfu* wk1b = (bfu*)take(2097152);
  bfu* wv1b = (bfu*)take(2097152); bfu* wo1b = (bfu*)take(2097152);
  bfu* wq2b = (bfu*)take(2097152); bfu* wk2b = (bfu*)take(2097152);
  bfu* wv2b = (bfu*)take(2097152); bfu* wo2b = (bfu*)take(2097152);
  bfu* c1wb = (bfu*)take(4194304);
  bfu* wtap = (bfu*)take(18874368);
  bfu* f1wb = (bfu*)take(4194304);
  bfu* f2wb = (bfu*)take(4194304);
  float* owin = dctb;
  bfu*   hbuf = cat;
  float* fin  = y2;

  const int M = 32768;
  auto gemm = [&](const bfu* A, const bfu* W, const float* bias, const float* add,
                  float* of, bfu* ob, int N, int K, int amode, int dy, int dx, int om) {
    dim3 g(N / 128, M / 128);
    k_gemm<<<g, 256, 0, stream>>>(A, W, bias, add, of, ob, M, N, K, amode, dy, dx, om);
  };

  // ---- weight conversion f32 -> bf16
  k_f2bf<<<4096, 256, 0, stream>>>(wq1, wq1b, 1048576);
  k_f2bf<<<4096, 256, 0, stream>>>(wk1, wk1b, 1048576);
  k_f2bf<<<4096, 256, 0, stream>>>(wv1, wv1b, 1048576);
  k_f2bf<<<4096, 256, 0, stream>>>(wo1, wo1b, 1048576);
  k_f2bf<<<4096, 256, 0, stream>>>(wq2, wq2b, 1048576);
  k_f2bf<<<4096, 256, 0, stream>>>(wk2, wk2b, 1048576);
  k_f2bf<<<4096, 256, 0, stream>>>(wv2, wv2b, 1048576);
  k_f2bf<<<4096, 256, 0, stream>>>(wo2, wo2b, 1048576);
  k_f2bf<<<4096, 256, 0, stream>>>(c1w, c1wb, 2097152);
  k_f2bf<<<4096, 256, 0, stream>>>(f1w, f1wb, 2097152);
  k_f2bf<<<4096, 256, 0, stream>>>(f2w, f2wb, 2097152);
  k_taps<<<4096, 256, 0, stream>>>(c2w, wtap);

  // ---- DCT branch + LayerNorms + windowing
  k_dct<<<2048, 256, 0, stream>>>(feat, dctb);
  k_lnwin<<<4096, 256, 0, stream>>>(feat, dctb, n1w, n1b, n2w, n2b, fwb, swb);

  // ---- attention 1: q<-sw, k/v<-fw  -> stf (idct) into cat[:,0:1024]
  gemm(swb, wq1b, bq1, nullptr, nullptr, qb, 1024, 1024, 0, 0, 0, 1);
  gemm(fwb, wk1b, bk1, nullptr, nullptr, kb, 1024, 1024, 0, 0, 0, 1);
  gemm(fwb, wv1b, bv1, nullptr, nullptr, vb, 1024, 1024, 0, 0, 0, 1);
  k_attn<<<4096, 256, 0, stream>>>(qb, kb, vb, tb1, ctx);
  gemm(ctx, wo1b, bo1, nullptr, owin, nullptr, 1024, 1024, 0, 0, 0, 0);
  k_idct_cat<<<2048, 256, 0, stream>>>(owin, cat);

  // ---- attention 2: q<-fw, k/v<-sw  -> fts into cat[:,1024:2048]
  gemm(fwb, wq2b, bq2, nullptr, nullptr, qb, 1024, 1024, 0, 0, 0, 1);
  gemm(swb, wk2b, bk2, nullptr, nullptr, kb, 1024, 1024, 0, 0, 0, 1);
  gemm(swb, wv2b, bv2, nullptr, nullptr, vb, 1024, 1024, 0, 0, 0, 1);
  k_attn<<<4096, 256, 0, stream>>>(qb, kb, vb, tb2, ctx);
  gemm(ctx, wo2b, bo2, nullptr, owin, nullptr, 1024, 1024, 0, 0, 0, 0);
  k_win2cat<<<131072, 256, 0, stream>>>(owin, cat);

  // ---- conv1x1 (K=2048) -> zero-padded NHWC, then conv3x3 as 9 tap GEMMs
  k_zero16<<<8192, 256, 0, stream>>>(y1p, 35684352ll);
  gemm(cat, c1wb, c1b, nullptr, nullptr, y1p, 1024, 2048, 0, 0, 0, 4);
  for (int tap = 0; tap < 9; ++tap) {
    int dy = tap / 3, dx = tap % 3;
    if (tap == 0)
      gemm(y1p, wtap + (size_t)tap * 1048576, c2b, nullptr, y2, nullptr, 1024, 1024, 1, dy, dx, 0);
    else
      gemm(y1p, wtap + (size_t)tap * 1048576, nullptr, y2, y2, nullptr, 1024, 1024, 1, dy, dx, 3);
  }

  // ---- BN (batch stats) + ReLU + residual with feat
  k_bnstats<<<1024, 256, 0, stream>>>(y2, bmean, bistd);
  k_bnres<<<131072, 256, 0, stream>>>(y2, bmean, bistd, bng, bnb, feat, res);

  // ---- MLP: LN -> fc1(gelu) -> fc2(+res)
  k_ln3<<<4096, 256, 0, stream>>>(res, n3w, n3b, tnb);
  gemm(tnb, f1wb, f1b, nullptr, nullptr, hbuf, 2048, 1024, 0, 0, 0, 2);
  gemm(hbuf, f2wb, f2b, res, fin, nullptr, 1024, 2048, 0, 0, 0, 3);

  // ---- NHWC -> NCHW output
  k_hwc2chw<<<131072, 256, 0, stream>>>(fin, (float*)d_out);
}
